// Net_int_2Edges_pool_54674933678648
// MI455X (gfx1250) — matrix operand order repaired and checked by reading, not compiled, not run on hardware
//
#include <hip/hip_runtime.h>
#include <hip/hip_bf16.h>

// ---------------------------------------------------------------------------
// CDNA5 (gfx1250) implementation of the NNConv/GRU/Set2Set graph network.
// Dense matmuls -> v_wmma_f32_16x16x32_f16 (wave32, f32 accumulate).
// B panels staged into LDS with the Tensor Data Mover (tensor_load_to_lds,
// s_wait_tensorcnt) when available.
// ---------------------------------------------------------------------------

typedef __attribute__((ext_vector_type(16))) _Float16 v16h;
typedef __attribute__((ext_vector_type(8)))  float    v8f;
typedef __attribute__((ext_vector_type(4)))  unsigned int u32x4;
typedef __attribute__((ext_vector_type(8)))  int i32x8;
typedef __attribute__((ext_vector_type(4)))  int i32x4;

#if defined(__has_builtin)
#if __has_builtin(__builtin_amdgcn_tensor_load_to_lds) && \
    __has_builtin(__builtin_amdgcn_s_wait_tensorcnt)
#define HAVE_TDM 1
#endif
#endif
#ifndef HAVE_TDM
#define HAVE_TDM 0
#endif

#define RR_SLOPE 0.2291666666666667f   // 11/48
#define BN_EPS   1e-5f

__device__ __forceinline__ float rrelu(float v) { return v >= 0.f ? v : RR_SLOPE * v; }
__device__ __forceinline__ float sigm(float v)  { return 1.f / (1.f + __expf(-v)); }

// ---- WMMA fragment loaders (CDNA5 16-bit layouts, CDNA5 ISA §7.12.2) ------
__device__ __forceinline__ v16h load_a_f16(const _Float16* A, int lda) {
  int lane = threadIdx.x & 31;
  int m = lane & 15, hh = lane >> 4;
  v16h a;
#pragma unroll
  for (int t = 0; t < 16; ++t) {
    int r = t >> 1, p = t & 1;
    int k = (r < 4) ? (2 * r + 8 * hh + p) : (16 + 2 * (r - 4) + 8 * hh + p);
    a[t] = A[m * lda + k];
  }
  return a;
}
__device__ __forceinline__ v16h load_b_f16(const _Float16* B, int ldb) {
  int lane = threadIdx.x & 31;
  int n = lane & 15, hh = lane >> 4;
  v16h b;
#pragma unroll
  for (int t = 0; t < 16; ++t) {
    int r = t >> 1, p = t & 1;
    int k = (r < 4) ? (2 * r + 8 * hh + p) : (16 + 2 * (r - 4) + 8 * hh + p);
    b[t] = B[k * ldb + n];
  }
  return b;
}
__device__ __forceinline__ v8f wmma_f16(v16h a, v16h b, v8f c) {
  return __builtin_amdgcn_wmma_f32_16x16x32_f16(false, a, false, b, (short)0, c,
                                                false, false);
}

// ---- Tensor Data Mover: async 2D f16 panel load into LDS -------------------
// D# per CDNA5 ISA: group0 = {flags, lds_addr, gaddr,
// gaddr_hi|type}, group1 = dims/strides. tile == tensor (in-bounds panel),
// rows spaced by stride_elems (may exceed cols).
#if HAVE_TDM
__device__ __forceinline__ void tdm_load_f16_2d(unsigned lds_off, const _Float16* gptr,
                                                unsigned cols, unsigned rows,
                                                unsigned stride_elems) {
  unsigned long long ga = (unsigned long long)(size_t)gptr;
  u32x4 g0;
  g0.x = 1u;                                                  // count=1, user D#
  g0.y = lds_off;                                             // lds_addr (bytes)
  g0.z = (unsigned)ga;                                        // global_addr[31:0]
  g0.w = (unsigned)((ga >> 32) & 0x01FFFFFFu) | (2u << 30);   // addr[56:32] | type=2
  i32x8 g1;
  g1[0] = 1 << 16;                                            // data_size = 2 bytes
  g1[1] = (int)((cols & 0xFFFFu) << 16);                      // tensor_dim0[15:0]
  g1[2] = (int)((cols >> 16) & 0xFFFFu) | (int)((rows & 0xFFFFu) << 16);
  g1[3] = (int)((rows >> 16) & 0xFFFFu) | (int)((cols & 0xFFFFu) << 16);  // tile_dim0
  g1[4] = (int)(rows & 0xFFFFu);                              // tile_dim1, tile_dim2=0
  g1[5] = (int)stride_elems;                                  // tensor_dim0_stride lo32
  g1[6] = 0;
  g1[7] = 0;
  i32x4 z4 = {0, 0, 0, 0};
#if __clang_major__ >= 23
  i32x8 z8 = {0, 0, 0, 0, 0, 0, 0, 0};
  __builtin_amdgcn_tensor_load_to_lds(g0, g1, z4, z4, z8, 0);
#else
  __builtin_amdgcn_tensor_load_to_lds(g0, g1, z4, z4, 0);
#endif
}
#endif

// ---- weight repack kernels -------------------------------------------------
__global__ void conv_copy_f16(_Float16* dst, const float* src, int n) {
  int i = blockIdx.x * 256 + threadIdx.x;
  if (i < n) dst[i] = (_Float16)src[i];
}
__global__ void conv_transpose_f16(_Float16* dst, const float* src, int rows, int cols) {
  int i = blockIdx.x * 256 + threadIdx.x;
  if (i < rows * cols) {
    int r = i / cols, c = i % cols;
    dst[c * rows + r] = (_Float16)src[i];
  }
}

__global__ void zero_f32(float* p, int n) {
  int i = blockIdx.x * 256 + threadIdx.x;
  if (i < n) p[i] = 0.f;
}

// ---- column mean / rsqrt(var+eps) over axis 0 ------------------------------
__global__ void colstats(const float* __restrict__ src, int M, int C,
                         float* mean, float* inv) {
  __shared__ float s1[256], s2[256];
  int c = blockIdx.x, tid = threadIdx.x;
  float a = 0.f, q = 0.f;
  for (int r = tid; r < M; r += 256) {
    float v = src[r * C + c];
    a += v; q += v * v;
  }
  s1[tid] = a; s2[tid] = q; __syncthreads();
  for (int st = 128; st > 0; st >>= 1) {
    if (tid < st) { s1[tid] += s1[tid + st]; s2[tid] += s2[tid + st]; }
    __syncthreads();
  }
  if (tid == 0) {
    float m = s1[0] / (float)M;
    float v = s2[0] / (float)M - m * m;
    mean[c] = m;
    inv[c]  = rsqrtf(v + BN_EPS);
  }
}

__global__ void bn_apply(const float* src, const float* mean, const float* inv,
                         const float* g, const float* b, float* dst, int C, int total) {
  int i = blockIdx.x * 256 + threadIdx.x;
  if (i >= total) return;
  int c = i % C;
  dst[i] = (src[i] - mean[c]) * inv[c] * g[c] + b[c];
}

// ---- node encoder: rrelu(BN(x) @ Wn + bnode) (K=8, scalar) -----------------
__global__ void node_encoder(const float* __restrict__ x, const float* mean8,
                             const float* inv8, const float* g, const float* b,
                             const float* __restrict__ Wn, const float* bnode,
                             float* out) {
  int i = blockIdx.x * 256 + threadIdx.x;
  int nrow = i >> 6, j = i & 63;
  float acc = bnode[j];
#pragma unroll
  for (int k = 0; k < 8; ++k) {
    float xv = (x[nrow * 8 + k] - mean8[k]) * inv8[k] * g[k] + b[k];
    acc += xv * Wn[k * 64 + j];
  }
  out[i] = rrelu(acc);
}

// ---- edge encoder: rrelu(edge_attr @ We + be), 19 -> 12 --------------------
__global__ void edge_encoder(const float* __restrict__ ea_in, const float* __restrict__ We,
                             const float* be, float* ea_out) {
  int i = blockIdx.x * 256 + threadIdx.x;
  int e = i / 12, k = i % 12;
  float acc = be[k];
#pragma unroll
  for (int t = 0; t < 19; ++t) acc += ea_in[e * 19 + t] * We[t * 12 + k];
  ea_out[i] = rrelu(acc);
}

// ---- destination-degree counts ---------------------------------------------
template <int HALF>
__global__ void count_edges(const int* __restrict__ eidx, float* cnt, int Etot) {
  int e = blockIdx.x * 256 + threadIdx.x;
  if (e >= Etot) return;
  int dstn = HALF ? eidx[(e + HALF) & (2 * HALF - 1)] : eidx[Etot + e];
  atomicAdd(&cnt[dstn], 1.0f);
}

// ---- NNConv via WMMA -------------------------------------------------------
// Per block (1 wave): 16 edges. A'[e, k*D+i] = ea[e,k]*x[src,i] in LDS (f16);
// multiply against Bp[(k*D+i), o] = Wnn[k, i*D+o]; TDM-stage each 16-wide B
// column panel; scatter-add 16xD rows to acc[dst].
template <int D, int F, int HALF>
__global__ void nnconv_wmma(const float* __restrict__ xin, const int* __restrict__ eidx,
                            const float* __restrict__ efeat,
                            const _Float16* __restrict__ Bp, float* acc, int Etot) {
  constexpr int KK  = F * D;
  constexpr int DSH = (D == 64) ? 6 : 7;
  __shared__ _Float16 smem[16 * KK + KK * 16];
  _Float16* As = smem;            // LDS offset 0
  _Float16* Bs = smem + 16 * KK;  // LDS offset 16*KK*2 bytes
  int lane = threadIdx.x;
  int ebase = blockIdx.x * 16;
  for (int el = 0; el < 16; ++el) {
    int e = ebase + el;
    const float* xr = xin + eidx[e] * D;
    const float* er = efeat + (HALF ? (e & (HALF - 1)) : e) * F;
    for (int idx = lane; idx < KK; idx += 32)
      As[el * KK + idx] = (_Float16)(er[idx >> DSH] * xr[idx & (D - 1)]);
  }
  __syncthreads();
  int n = lane & 15, mh = (lane >> 4) * 8;
  for (int nt = 0; nt < D / 16; ++nt) {
#if HAVE_TDM
    __syncthreads();   // previous panel's DS reads retired
    tdm_load_f16_2d((unsigned)(16 * KK * 2), Bp + nt * 16, 16, KK, D);
    __builtin_amdgcn_s_wait_tensorcnt(0);
    asm volatile("" ::: "memory");
#else
    __syncthreads();
    for (int t = lane; t < KK * 16; t += 32)
      Bs[t] = Bp[(t >> 4) * D + nt * 16 + (t & 15)];
    __syncthreads();
#endif
    v8f c = {};
#pragma unroll 4
    for (int kc = 0; kc < KK; kc += 32) {
      v16h a = load_a_f16(As + kc, KK);
      v16h b = load_b_f16(Bs + kc * 16, 16);
      c = wmma_f16(a, b, c);
    }
#pragma unroll
    for (int r = 0; r < 8; ++r) {
      int e = ebase + mh + r;
      int dstn = HALF ? eidx[(e + HALF) & (2 * HALF - 1)] : eidx[Etot + e];
      atomicAdd(&acc[dstn * D + nt * 16 + n], c[r]);
    }
  }
}

__global__ void conv_finalize(const float* acc, const float* cnt, const float* bias,
                              float* m, int dsh) {
  int i = blockIdx.x * 256 + threadIdx.x;
  int D = 1 << dsh;
  int nrow = i >> dsh, o = i & (D - 1);
  float c = cnt[nrow]; if (c < 1.f) c = 1.f;
  m[i] = rrelu(acc[i] / c + bias[o]);
}

// ---- generic WMMA GEMM: C[M,N] = A[M,K](f32) * B[K,N](f16) -----------------
// One wave per 16x64 tile: 4 accumulators share one A fragment. B panel
// (K x 64) TDM-staged in LDS. flags bit0: rrelu, bit1: accumulate into C.
template <int K>
__global__ void gemm_wmma(const float* __restrict__ A, int lda,
                          const _Float16* __restrict__ B, int ldb,
                          float* C, int ldc, const float* bias, int flags) {
  __shared__ _Float16 smem[16 * 32 + K * 64];
  _Float16* As = smem;            // LDS offset 0
  _Float16* Bs = smem + 16 * 32;  // LDS offset 1024 bytes
  int lane = threadIdx.x;
  int mb = blockIdx.x * 16, nb = blockIdx.y * 64;
#if HAVE_TDM
  tdm_load_f16_2d(1024u, B + nb, 64, K, ldb);
#else
  for (int t = lane; t < K * 64; t += 32)
    Bs[t] = B[(t >> 6) * ldb + nb + (t & 63)];
#endif
  int n = lane & 15, mh = (lane >> 4) * 8;
  v8f c0 = {}, c1 = {}, c2 = {}, c3 = {};
  if (flags & 2) {
#pragma unroll
    for (int r = 0; r < 8; ++r) {
      int ro = (mb + mh + r) * ldc + nb + n;
      c0[r] = C[ro]; c1[r] = C[ro + 16]; c2[r] = C[ro + 32]; c3[r] = C[ro + 48];
    }
  }
#if HAVE_TDM
  __builtin_amdgcn_s_wait_tensorcnt(0);
  asm volatile("" ::: "memory");
#else
  __syncthreads();
#endif
#pragma unroll
  for (int k0 = 0; k0 < K; k0 += 32) {
    __syncthreads();
    for (int t = lane; t < 512; t += 32)
      As[t] = (_Float16)A[(mb + (t >> 5)) * lda + k0 + (t & 31)];
    if (k0 + 32 < K) __builtin_prefetch(&A[mb * lda + k0 + 32], 0, 0);
    __syncthreads();
    v16h a = load_a_f16(As, 32);
    c0 = wmma_f16(a, load_b_f16(Bs + k0 * 64 +  0, 64), c0);
    c1 = wmma_f16(a, load_b_f16(Bs + k0 * 64 + 16, 64), c1);
    c2 = wmma_f16(a, load_b_f16(Bs + k0 * 64 + 32, 64), c2);
    c3 = wmma_f16(a, load_b_f16(Bs + k0 * 64 + 48, 64), c3);
  }
  int col = nb + n;
  float b0 = bias ? bias[col] : 0.f,      b1 = bias ? bias[col + 16] : 0.f;
  float b2 = bias ? bias[col + 32] : 0.f, b3 = bias ? bias[col + 48] : 0.f;
#pragma unroll
  for (int r = 0; r < 8; ++r) {
    int ro = (mb + mh + r) * ldc + col;
    float v0 = c0[r] + b0, v1 = c1[r] + b1, v2 = c2[r] + b2, v3 = c3[r] + b3;
    if (flags & 1) { v0 = rrelu(v0); v1 = rrelu(v1); v2 = rrelu(v2); v3 = rrelu(v3); }
    C[ro] = v0; C[ro + 16] = v1; C[ro + 32] = v2; C[ro + 48] = v3;
  }
}

// ---- GRU pointwise ---------------------------------------------------------
__global__ void gru_elem(const float* __restrict__ gi, const float* __restrict__ gh,
                         const float* bih, const float* bhh, float* h, int hsh) {
  int i = blockIdx.x * 256 + threadIdx.x;
  int H = 1 << hsh;
  int m = i >> hsh, j = i & (H - 1);
  int b3 = m * 3 * H;
  float r  = sigm(gi[b3 + j] + bih[j] + gh[b3 + j] + bhh[j]);
  float z  = sigm(gi[b3 + H + j] + bih[H + j] + gh[b3 + H + j] + bhh[H + j]);
  float nn = tanhf(gi[b3 + 2 * H + j] + bih[2 * H + j] +
                   r * (gh[b3 + 2 * H + j] + bhh[2 * H + j]));
  float hv = h[i];
  h[i] = (1.f - z) * nn + z * hv;
}

// ---- Set2Set LSTM pointwise (G=256, H=128) ---------------------------------
__global__ void lstm_elem(const float* __restrict__ gates, const float* bih,
                          const float* bhh, float* cstate, float* hstate) {
  int i = blockIdx.x * 256 + threadIdx.x;
  int g = i >> 7, j = i & 127;
  int b4 = g * 512;
  float iv = sigm(gates[b4 + j] + bih[j] + bhh[j]);
  float fv = sigm(gates[b4 + 128 + j] + bih[128 + j] + bhh[128 + j]);
  float gv = tanhf(gates[b4 + 256 + j] + bih[256 + j] + bhh[256 + j]);
  float ov = sigm(gates[b4 + 384 + j] + bih[384 + j] + bhh[384 + j]);
  float cv = fv * cstate[i] + iv * gv;
  cstate[i] = cv;
  hstate[i] = ov * tanhf(cv);
}

// ---- Set2Set attention: 32 contiguous nodes per graph ----------------------
__global__ void s2s_attend(const float* __restrict__ xfeat, const float* __restrict__ hstate,
                           float* qstar) {
  __shared__ float sa[32];
  int g = blockIdx.x, lane = threadIdx.x;
  int node = g * 32 + lane;
  float e = 0.f;
  for (int c = 0; c < 128; ++c) e += xfeat[node * 128 + c] * hstate[g * 128 + c];
  sa[lane] = e; __syncthreads();
  float mx = sa[0];
  for (int t = 1; t < 32; ++t) mx = fmaxf(mx, sa[t]);
  __syncthreads();
  float ex = __expf(e - mx);
  sa[lane] = ex; __syncthreads();
  float den = 0.f;
  for (int t = 0; t < 32; ++t) den += sa[t];
  __syncthreads();
  sa[lane] = ex / den; __syncthreads();
  for (int c = lane; c < 128; c += 32) {
    float r = 0.f;
    for (int t = 0; t < 32; ++t) r += sa[t] * xfeat[(g * 32 + t) * 128 + c];
    qstar[g * 256 + 128 + c] = r;
    qstar[g * 256 + c] = hstate[g * 128 + c];
  }
}

// ---- output head -----------------------------------------------------------
__global__ void yhat_build(const float* __restrict__ out2, const int* __restrict__ ei3,
                           const int* __restrict__ batch, const float* __restrict__ qstar,
                           float* yhat) {
  int e = blockIdx.x, tid = threadIdx.x;
  int i0 = ei3[e], i1 = ei3[1024 + e];
  int g = batch[i0];
  for (int c = tid; c < 640; c += 128) {
    float v;
    if (c < 128)      { float a = out2[i0 * 128 + c], b = out2[i1 * 128 + c]; v = 0.5f * (a + b); }
    else if (c < 256) { int j = c - 128; v = out2[i0 * 128 + j] * out2[i1 * 128 + j]; }
    else if (c < 384) { int j = c - 256; float d = out2[i0 * 128 + j] - out2[i1 * 128 + j]; v = d * d; }
    else              { v = qstar[g * 256 + (c - 384)]; }
    yhat[e * 640 + c] = v;
  }
}

__global__ void wmat_build(const float* __restrict__ ea3, const float* __restrict__ Ww,
                           float* wmat) {
  int i = blockIdx.x * 256 + threadIdx.x;
  if (i >= 1024 * 640) return;
  int e = i / 640, c = i % 640;
  float s = 0.f;
#pragma unroll
  for (int t = 0; t < 8; ++t) s += ea3[e * 8 + t] * Ww[t * 640 + c];
  wmat[i] = s;
}
__global__ void bvec_build(const float* __restrict__ ea3, const float* __restrict__ Wb,
                           float* bvec) {
  int e = blockIdx.x * 256 + threadIdx.x;
  if (e >= 1024) return;
  float s = 0.f;
#pragma unroll
  for (int t = 0; t < 8; ++t) s += ea3[e * 8 + t] * Wb[t];
  bvec[e] = s;
}

__global__ void final_reduce(const float* __restrict__ yhat, const float* meanY,
                             const float* invY, const float* g, const float* b,
                             const float* __restrict__ wmat, const float* bvec,
                             float* outp) {
  __shared__ float sm[256];
  int e = blockIdx.x, tid = threadIdx.x;
  float s = 0.f;
  for (int c = tid; c < 640; c += 256) {
    float yn = (yhat[e * 640 + c] - meanY[c]) * invY[c] * g[c] + b[c];
    s += yn * wmat[e * 640 + c];
  }
  sm[tid] = s; __syncthreads();
  for (int st = 128; st > 0; st >>= 1) {
    if (tid < st) sm[tid] += sm[tid + st];
    __syncthreads();
  }
  if (tid == 0) outp[e] = sm[0] + bvec[e];
}

// ---------------------------------------------------------------------------
extern "C" void kernel_launch(void* const* d_in, const int* in_sizes, int n_in,
                              void* d_out, int out_size, void* d_ws, size_t ws_size,
                              hipStream_t stream) {
  (void)in_sizes; (void)n_in; (void)out_size; (void)ws_size;
  const int N = 8192, E = 16384, E3 = 1024, E3S = 2048, G = 256;

  const float* x     = (const float*)d_in[0];
  const int*   ei    = (const int*)  d_in[1];
  const float* eatt  = (const float*)d_in[2];
  const int*   ei3   = (const int*)  d_in[3];
  const float* ea3   = (const float*)d_in[4];
  const int*   batch = (const int*)  d_in[5];
  const float* bnx_g = (const float*)d_in[6];
  const float* bnx_b = (const float*)d_in[7];
  const float* Wn    = (const float*)d_in[8];
  const float* bnode = (const float*)d_in[9];
  const float* We    = (const float*)d_in[10];
  const float* be    = (const float*)d_in[11];
  const float* Wnn1  = (const float*)d_in[12];
  const float* bias1 = (const float*)d_in[13];
  const float* Wih1  = (const float*)d_in[14];
  const float* Whh1  = (const float*)d_in[15];
  const float* bih1  = (const float*)d_in[16];
  const float* bhh1  = (const float*)d_in[17];
  const float* bnc_g = (const float*)d_in[18];
  const float* bnc_b = (const float*)d_in[19];
  const float* Wc1   = (const float*)d_in[20];
  const float* bc1   = (const float*)d_in[21];
  const float* Wc2   = (const float*)d_in[22];
  const float* bc2   = (const float*)d_in[23];
  const float* Wnn2  = (const float*)d_in[24];
  const float* bias2 = (const float*)d_in[25];
  const float* Wih2  = (const float*)d_in[26];
  const float* Whh2  = (const float*)d_in[27];
  const float* bih2  = (const float*)d_in[28];
  const float* bhh2  = (const float*)d_in[29];
  const float* Wsih  = (const float*)d_in[30];
  const float* Wshh  = (const float*)d_in[31];
  const float* bsih  = (const float*)d_in[32];
  const float* bshh  = (const float*)d_in[33];
  const float* bno_g = (const float*)d_in[34];
  const float* bno_b = (const float*)d_in[35];
  const float* Ww    = (const float*)d_in[36];
  const float* Wb    = (const float*)d_in[37];
  float* outp = (float*)d_out;

  // ---- workspace carve-up --------------------------------------------------
  char* p = (char*)d_ws;
  auto alloc = [&](size_t elems, size_t esz) -> void* {
    void* r = (void*)p;
    p += (elems * esz + 255) & ~(size_t)255;
    return r;
  };
  _Float16* B1p    = (_Float16*)alloc(768 * 64, 2);
  _Float16* Wih1T  = (_Float16*)alloc(64 * 192, 2);
  _Float16* Whh1T  = (_Float16*)alloc(64 * 192, 2);
  _Float16* Wc1h   = (_Float16*)alloc(64 * 128, 2);
  _Float16* Wc2h   = (_Float16*)alloc(128 * 128, 2);
  _Float16* B2p    = (_Float16*)alloc(1024 * 128, 2);
  _Float16* Wih2T  = (_Float16*)alloc(128 * 384, 2);
  _Float16* Whh2T  = (_Float16*)alloc(128 * 384, 2);
  _Float16* WsihT  = (_Float16*)alloc(256 * 512, 2);
  _Float16* WshhT  = (_Float16*)alloc(128 * 512, 2);
  float* mean8 = (float*)alloc(8, 4);
  float* inv8  = (float*)alloc(8, 4);
  float* meanC = (float*)alloc(64, 4);
  float* invC  = (float*)alloc(64, 4);
  float* meanY = (float*)alloc(640, 4);
  float* invY  = (float*)alloc(640, 4);
  float* cur1  = (float*)alloc((size_t)N * 64, 4);
  float* m1    = (float*)alloc((size_t)N * 64, 4);
  float* bnout = (float*)alloc((size_t)N * 64, 4);
  float* gi    = (float*)alloc((size_t)N * 384, 4);
  float* gh    = (float*)alloc((size_t)N * 384, 4);
  float* eab   = (float*)alloc((size_t)E * 12, 4);
  float* acc   = (float*)alloc((size_t)N * 128, 4);
  float* cnt   = (float*)alloc(N, 4);
  float* c1b   = (float*)alloc((size_t)N * 128, 4);
  float* cur2  = (float*)alloc((size_t)N * 128, 4);
  float* m2    = (float*)alloc((size_t)N * 128, 4);
  float* qstar = (float*)alloc((size_t)G * 256, 4);
  float* hs    = (float*)alloc((size_t)G * 128, 4);
  float* cs    = (float*)alloc((size_t)G * 128, 4);
  float* gates = (float*)alloc((size_t)G * 512, 4);
  float* yhat  = (float*)alloc((size_t)E3 * 640, 4);
  float* wmat  = (float*)alloc((size_t)E3 * 640, 4);
  float* bvec  = (float*)alloc(E3, 4);

  auto cdiv = [](int a, int b) { return (a + b - 1) / b; };

  // ---- pack weights to f16 -------------------------------------------------
  conv_copy_f16<<<cdiv(768 * 64, 256), 256, 0, stream>>>(B1p, Wnn1, 768 * 64);
  conv_transpose_f16<<<cdiv(192 * 64, 256), 256, 0, stream>>>(Wih1T, Wih1, 192, 64);
  conv_transpose_f16<<<cdiv(192 * 64, 256), 256, 0, stream>>>(Whh1T, Whh1, 192, 64);
  conv_copy_f16<<<cdiv(64 * 128, 256), 256, 0, stream>>>(Wc1h, Wc1, 64 * 128);
  conv_copy_f16<<<cdiv(128 * 128, 256), 256, 0, stream>>>(Wc2h, Wc2, 128 * 128);
  conv_copy_f16<<<cdiv(1024 * 128, 256), 256, 0, stream>>>(B2p, Wnn2, 1024 * 128);
  conv_transpose_f16<<<cdiv(384 * 128, 256), 256, 0, stream>>>(Wih2T, Wih2, 384, 128);
  conv_transpose_f16<<<cdiv(384 * 128, 256), 256, 0, stream>>>(Whh2T, Whh2, 384, 128);
  conv_transpose_f16<<<cdiv(512 * 256, 256), 256, 0, stream>>>(WsihT, Wsih, 512, 256);
  conv_transpose_f16<<<cdiv(512 * 128, 256), 256, 0, stream>>>(WshhT, Wshh, 512, 128);

  // ---- stage 0: encoders ---------------------------------------------------
  colstats<<<8, 256, 0, stream>>>(x, N, 8, mean8, inv8);
  node_encoder<<<N * 64 / 256, 256, 0, stream>>>(x, mean8, inv8, bnx_g, bnx_b, Wn,
                                                 bnode, cur1);
  edge_encoder<<<E * 12 / 256, 256, 0, stream>>>(eatt, We, be, eab);

  // ---- stage 1: NNConv(64) + GRU, 2 iterations -----------------------------
  zero_f32<<<cdiv(N, 256), 256, 0, stream>>>(cnt, N);
  count_edges<0><<<cdiv(E, 256), 256, 0, stream>>>(ei, cnt, E);
  for (int it = 0; it < 2; ++it) {
    zero_f32<<<cdiv(N * 64, 256), 256, 0, stream>>>(acc, N * 64);
    nnconv_wmma<64, 12, 0><<<E / 16, 32, 0, stream>>>(cur1, ei, eab, B1p, acc, E);
    conv_finalize<<<N * 64 / 256, 256, 0, stream>>>(acc, cnt, bias1, m1, 6);
    gemm_wmma<64><<<dim3(N / 16, 192 / 64), 32, 0, stream>>>(m1, 64, Wih1T, 192, gi,
                                                             192, nullptr, 0);
    gemm_wmma<64><<<dim3(N / 16, 192 / 64), 32, 0, stream>>>(cur1, 64, Whh1T, 192, gh,
                                                             192, nullptr, 0);
    gru_elem<<<N * 64 / 256, 256, 0, stream>>>(gi, gh, bih1, bhh1, cur1, 6);
  }

  // ---- BN + MLP 64 -> 128 -> 128 -------------------------------------------
  colstats<<<64, 256, 0, stream>>>(cur1, N, 64, meanC, invC);
  bn_apply<<<cdiv(N * 64, 256), 256, 0, stream>>>(cur1, meanC, invC, bnc_g, bnc_b,
                                                  bnout, 64, N * 64);
  gemm_wmma<64><<<dim3(N / 16, 128 / 64), 32, 0, stream>>>(bnout, 64, Wc1h, 128, c1b,
                                                           128, bc1, 1);
  gemm_wmma<128><<<dim3(N / 16, 128 / 64), 32, 0, stream>>>(c1b, 128, Wc2h, 128, cur2,
                                                            128, bc2, 1);

  // ---- stage 2: symmetric NNConv(128) + GRU, 2 iterations ------------------
  zero_f32<<<cdiv(N, 256), 256, 0, stream>>>(cnt, N);
  count_edges<1024><<<cdiv(E3S, 256), 256, 0, stream>>>(ei3, cnt, E3S);
  for (int it = 0; it < 2; ++it) {
    zero_f32<<<cdiv(N * 128, 256), 256, 0, stream>>>(acc, N * 128);
    nnconv_wmma<128, 8, 1024><<<E3S / 16, 32, 0, stream>>>(cur2, ei3, ea3, B2p, acc, E3S);
    conv_finalize<<<N * 128 / 256, 256, 0, stream>>>(acc, cnt, bias2, m2, 7);
    gemm_wmma<128><<<dim3(N / 16, 384 / 64), 32, 0, stream>>>(m2, 128, Wih2T, 384, gi,
                                                              384, nullptr, 0);
    gemm_wmma<128><<<dim3(N / 16, 384 / 64), 32, 0, stream>>>(cur2, 128, Whh2T, 384, gh,
                                                              384, nullptr, 0);
    gru_elem<<<N * 128 / 256, 256, 0, stream>>>(gi, gh, bih2, bhh2, cur2, 7);
  }

  // ---- Set2Set pooling (3 steps) -------------------------------------------
  zero_f32<<<cdiv(G * 256, 256), 256, 0, stream>>>(qstar, G * 256);
  zero_f32<<<cdiv(G * 128, 256), 256, 0, stream>>>(hs, G * 128);
  zero_f32<<<cdiv(G * 128, 256), 256, 0, stream>>>(cs, G * 128);
  for (int s = 0; s < 3; ++s) {
    gemm_wmma<256><<<dim3(G / 16, 512 / 64), 32, 0, stream>>>(qstar, 256, WsihT, 512,
                                                              gates, 512, nullptr, 0);
    gemm_wmma<128><<<dim3(G / 16, 512 / 64), 32, 0, stream>>>(hs, 128, WshhT, 512,
                                                              gates, 512, nullptr, 2);
    lstm_elem<<<G * 128 / 256, 256, 0, stream>>>(gates, bsih, bshh, cs, hs);
    s2s_attend<<<G, 32, 0, stream>>>(cur2, hs, qstar);
  }

  // ---- output head ---------------------------------------------------------
  yhat_build<<<E3, 128, 0, stream>>>(cur2, ei3, batch, qstar, yhat);
  colstats<<<640, 256, 0, stream>>>(yhat, E3, 640, meanY, invY);
  wmat_build<<<cdiv(E3 * 640, 256), 256, 0, stream>>>(ea3, Ww, wmat);
  bvec_build<<<cdiv(E3, 256), 256, 0, stream>>>(ea3, Wb, bvec);
  final_reduce<<<E3, 256, 0, stream>>>(yhat, meanY, invY, bno_g, bno_b, wmat, bvec, outp);
}
